// WaveNetModel_90151363543222
// MI455X (gfx1250) — compile-verified, matrix operand order checked by reading, not run on hardware
//
#include <hip/hip_runtime.h>
#include <cstddef>

// ===========================================================================
// WaveNet forward on MI455X (gfx1250, wave32, WMMA f16 -> f32 accum).
//
// Pipeline (all on `stream`):
//   memset(total=0)
//   k_cvt_weights : fp32 weights -> f16 row-major [M][K] tables in ws
//   k_causal      : one-hot conv == gather:  h[b,r,t] = cw[r,x[t],0]+cw[r,x[t+1],1]+cb
//   per layer i (20x):
//     k_gated    : GEMM  Wfg(128x128) x X(128x64/tile) -> tanh*sigmoid -> out16
//     k_res_skip : GEMM  W2 (320x 64) x out(64x64/tile) -> h += dense, total += skip
//   k_post        : relu -> GEMM(256x256) -> relu -> GEMM(256x256) -> d_out
//
// Workspace (~170 MB): h[8][64][16383] f32 | out16[8][16448][64] f16 |
//                      total[8][256][14337] f32 | f16 weight tables (~1.7 MB)
// ===========================================================================

#define DEV_INLINE __device__ __forceinline__

typedef _Float16 v8h  __attribute__((ext_vector_type(8)));
typedef _Float16 v16h __attribute__((ext_vector_type(16)));
typedef float    v8f  __attribute__((ext_vector_type(8)));

constexpr int kB      = 8;
constexpr int kT      = 16384;
constexpr int kT1     = 16383;   // length of h after causal conv
constexpr int kOutW   = 14337;
constexpr int kL      = 20;
constexpr int kOutRows = 16448;  // padded time rows for out16 (>= 16383 + 64)

// ---- workspace layout (all offsets multiples of 256 bytes) ----------------
constexpr size_t H_BYTES   = (size_t)kB * 64 * kT1 * 4;        //  33,552,384
constexpr size_t O16_BYTES = (size_t)kB * kOutRows * 64 * 2;   //  16,842,752
constexpr size_t TOT_BYTES = (size_t)kB * 256 * kOutW * 4;     // 117,448,704
constexpr size_t WFG_HALF  = (size_t)kL * 128 * 128;           // 327,680
constexpr size_t W2_HALF   = (size_t)kL * 320 * 64;            // 409,600
constexpr size_t WP_HALF   = (size_t)256 * 256;                //  65,536

DEV_INLINE v8f vzero8() { v8f z = {0.f,0.f,0.f,0.f,0.f,0.f,0.f,0.f}; return z; }

// Load a 16x32 f16 fragment (A: rows=M, B: rows=N) from a row-major [rows][ld]
// matrix.  Per CDNA5 layout: lane L holds row (L&15); lanes 16-31 start at
// K=+8; each lane's 16 halves are K = kb+{0..7} and kb+16+{0..7}  ==
// two contiguous 16-byte loads per lane.
DEV_INLINE v16h ld_frag(const _Float16* base, int ld, int lane) {
  const int row = lane & 15;
  const int kb  = (lane >> 4) << 3;
  const _Float16* p = base + row * ld + kb;
  v8h lo = *reinterpret_cast<const v8h*>(p);
  v8h hi = *reinterpret_cast<const v8h*>(p + 16);
  return __builtin_shufflevector(lo, hi,
      0,1,2,3,4,5,6,7,8,9,10,11,12,13,14,15);
}

DEV_INLINE v8f wmma16(v16h a, v16h b, v8f c) {
  return __builtin_amdgcn_wmma_f32_16x16x32_f16(
      false, a, false, b, (short)0, c, false, false);
}

// ---------------------------------------------------------------------------
// Weight conversion: fp32 -> f16 row-major [M][K] GEMM tables.
// ---------------------------------------------------------------------------
__global__ __launch_bounds__(256) void k_cvt_weights(
    const float* __restrict__ filt_w, const float* __restrict__ gate_w,
    const float* __restrict__ dense_w, const float* __restrict__ skip_w,
    const float* __restrict__ pp1_w, const float* __restrict__ pp2_w,
    _Float16* __restrict__ wfg, _Float16* __restrict__ w2,
    _Float16* __restrict__ wp1, _Float16* __restrict__ wp2)
{
  const size_t idx = (size_t)blockIdx.x * 256 + threadIdx.x;
  if (idx < WFG_HALF) {
    // Wfg[i][m][k]: m<64 filt, m>=64 gate; k = r + 64*tap
    int i = (int)(idx / (128 * 128)), rem = (int)(idx % (128 * 128));
    int m = rem >> 7, k = rem & 127, r = k & 63, tap = k >> 6;
    float v = (m < 64) ? filt_w[(((size_t)i * 64 + m) * 64 + r) * 2 + tap]
                       : gate_w[(((size_t)i * 64 + (m - 64)) * 64 + r) * 2 + tap];
    wfg[idx] = (_Float16)v;
  } else if (idx < WFG_HALF + W2_HALF) {
    size_t j = idx - WFG_HALF;
    int i = (int)(j / (320 * 64)), rem = (int)(j % (320 * 64));
    int m = rem >> 6, k = rem & 63;
    float v = (m < 64) ? dense_w[((size_t)i * 64 + m) * 64 + k]
                       : skip_w[((size_t)i * 256 + (m - 64)) * 64 + k];
    w2[j] = (_Float16)v;
  } else if (idx < WFG_HALF + W2_HALF + WP_HALF) {
    size_t j = idx - WFG_HALF - W2_HALF;
    wp1[j] = (_Float16)pp1_w[j];
  } else if (idx < WFG_HALF + W2_HALF + 2 * WP_HALF) {
    size_t j = idx - WFG_HALF - W2_HALF - WP_HALF;
    wp2[j] = (_Float16)pp2_w[j];
  }
}

// ---------------------------------------------------------------------------
// Causal conv: one-hot(x) * W is a gather, no GEMM needed.
// ---------------------------------------------------------------------------
__global__ __launch_bounds__(256) void k_causal(
    const int* __restrict__ x, const float* __restrict__ cw,
    const float* __restrict__ cb, float* __restrict__ h)
{
  const int t = blockIdx.x * 256 + threadIdx.x;
  const int r = blockIdx.y;
  const int b = blockIdx.z;
  if (t < kT1) {
    int x0 = x[(size_t)b * kT + t];
    int x1 = x[(size_t)b * kT + t + 1];
    float v = cw[((size_t)r * 256 + x0) * 2 + 0] +
              cw[((size_t)r * 256 + x1) * 2 + 1] + cb[r];
    h[((size_t)b * 64 + r) * kT1 + t] = v;
  }
}

// ---------------------------------------------------------------------------
// K1: gated dilated conv.  One WG = one batch x 64 output timesteps.
// GEMM: Wfg(M=128: filt|gate, K=128: 64ch x 2 taps) x X(128 x 64).
// Each of 8 waves owns one 16-row M tile; 4 N tiles; 4 K steps => 16 WMMA.
// ---------------------------------------------------------------------------
__global__ __launch_bounds__(256) void k_gated(
    const float* __restrict__ h, const _Float16* __restrict__ wfg,
    const float* __restrict__ filt_b, const float* __restrict__ gate_b,
    _Float16* __restrict__ out16,
    int layer, int dil, int off_in, int len_out)
{
  __shared__ __align__(16) _Float16 X[64][136];   // [n][k], pad 136 (17x16B rows)
  __shared__ float FG[128][66];                   // f rows 0..63, g rows 64..127

  const int b    = blockIdx.y;
  const int toff = blockIdx.x * 64;
  const int tid  = threadIdx.x;
  const int lane = tid & 31;
  const int wv   = tid >> 5;
  const float* hb = h + (size_t)b * 64 * kT1;

  // Stage X[n][r + 64*tap] = f16(h[b][r][off_in + toff + n + tap*dil])
  {
    const int n  = tid & 63;           // lane-fastest -> coalesced h reads
    const int kq = tid >> 6;           // 0..3
    const int tg = toff + n;
    for (int j = 0; j < 32; ++j) {
      int k = kq * 32 + j;
      int r = k & 63, tap = k >> 6;
      float v = 0.f;
      if (tg < len_out)
        v = hb[(size_t)r * kT1 + off_in + tg + tap * dil];
      X[n][k] = (_Float16)v;
    }
  }
  __syncthreads();

  // GEMM (wave-uniform; EXEC all ones around every WMMA)
  {
    v8f acc[4] = {vzero8(), vzero8(), vzero8(), vzero8()};
    const _Float16* wrow = wfg + (size_t)layer * 128 * 128 + (size_t)(wv * 16) * 128;
    for (int ks = 0; ks < 128; ks += 32) {
      v16h a = ld_frag(wrow + ks, 128, lane);
#pragma unroll
      for (int nt = 0; nt < 4; ++nt) {
        v16h bf = ld_frag(&X[nt * 16][0] + ks, 136, lane);
        acc[nt] = wmma16(a, bf, acc[nt]);
      }
    }
    // Scatter accumulators to LDS so f (m) and g (m+64) can be paired.
    const int mrow = wv * 16 + 8 * (lane >> 4);
    const int ncol = lane & 15;
#pragma unroll
    for (int nt = 0; nt < 4; ++nt)
#pragma unroll
      for (int v = 0; v < 8; ++v)
        FG[mrow + v][nt * 16 + ncol] = acc[nt][v];
  }
  __syncthreads();

  // Activation: out = tanh(f+fb) * sigmoid(g+gb), store f16 time-major [t][64].
  {
    const int dch = tid & 63;
    const int nq  = tid >> 6;
    const float fb = filt_b[layer * 64 + dch];
    const float gb = gate_b[layer * 64 + dch];
    for (int j = 0; j < 16; ++j) {
      int n  = nq + 4 * j;
      int tg = toff + n;
      if (tg < len_out) {
        float f = FG[dch][n] + fb;
        float g = FG[dch + 64][n] + gb;
        float act = tanhf(f) * (1.f / (1.f + __expf(-g)));
        out16[((size_t)b * kOutRows + tg) * 64 + dch] = (_Float16)act;
      }
    }
  }
}

// ---------------------------------------------------------------------------
// K2: 1x1 dense + skip, fused residual.
// GEMM: W2(M=320: dense 64 | skip 256, K=64) x out(64 x 64/tile).
// h[t] is updated in place (unique element ownership -> no race / atomics);
// total[t - tail] accumulated in place (unique ownership per layer launch).
// ---------------------------------------------------------------------------
__global__ __launch_bounds__(256) void k_res_skip(
    float* __restrict__ h, const _Float16* __restrict__ out16,
    const _Float16* __restrict__ w2,
    const float* __restrict__ dense_b, const float* __restrict__ skip_b,
    float* __restrict__ total,
    int layer, int off_next, int len_out, int tail)
{
  const int b    = blockIdx.y;
  const int toff = blockIdx.x * 64;
  const int tid  = threadIdx.x;
  const int lane = tid & 31;
  const int wv   = tid >> 5;
  const _Float16* xb = out16 + (size_t)b * kOutRows * 64;

  // Preload all B fragments (2 K-steps x 4 N-tiles) straight from global:
  // out16 rows are 128B contiguous -> two b128 loads per lane per fragment.
  v16h bf[2][4];
#pragma unroll
  for (int ks = 0; ks < 2; ++ks)
#pragma unroll
    for (int nt = 0; nt < 4; ++nt)
      bf[ks][nt] = ld_frag(xb + (size_t)(toff + nt * 16) * 64 + ks * 32, 64, lane);

  const int nmt = (wv < 4) ? 3 : 2;      // waves 0-3: M tiles {w,w+8,w+16}
  for (int im = 0; im < nmt; ++im) {
    const int mt = wv + im * 8;
    v8f acc[4] = {vzero8(), vzero8(), vzero8(), vzero8()};
    const _Float16* wrow = w2 + (size_t)layer * 320 * 64 + (size_t)(mt * 16) * 64;
#pragma unroll
    for (int ks = 0; ks < 2; ++ks) {
      v16h a = ld_frag(wrow + ks * 32, 64, lane);
#pragma unroll
      for (int nt = 0; nt < 4; ++nt)
        acc[nt] = wmma16(a, bf[ks][nt], acc[nt]);
    }

    const int mbase = mt * 16 + 8 * (lane >> 4);
    const int ncol  = lane & 15;
    if (mt < 4) {            // dense rows -> residual h[off_next + n] += ...
#pragma unroll
      for (int v = 0; v < 8; ++v) {
        const int r = mbase + v;
        const float bias = dense_b[layer * 64 + r];
#pragma unroll
        for (int nt = 0; nt < 4; ++nt) {
          int n = toff + nt * 16 + ncol;
          if (n < len_out) {
            size_t idx = ((size_t)b * 64 + r) * kT1 + off_next + n;
            h[idx] += acc[nt][v] + bias;
          }
        }
      }
    } else {                 // skip rows -> total[n - tail] += ...
#pragma unroll
      for (int v = 0; v < 8; ++v) {
        const int s = mbase + v - 64;
        const float bias = skip_b[layer * 256 + s];
#pragma unroll
        for (int nt = 0; nt < 4; ++nt) {
          int n = toff + nt * 16 + ncol;
          if (n < len_out && n >= tail) {
            size_t idx = ((size_t)b * 256 + s) * kOutW + (n - tail);
            total[idx] += acc[nt][v] + bias;
          }
        }
      }
    }
  }
}

// ---------------------------------------------------------------------------
// K3: fused post-net: relu -> 1x1(256x256) -> relu -> 1x1(256x256).
// One WG = one batch x 64 timesteps; both GEMMs through LDS, no c1 buffer.
// ---------------------------------------------------------------------------
__global__ __launch_bounds__(256) void k_post(
    const float* __restrict__ total,
    const _Float16* __restrict__ wp1, const _Float16* __restrict__ wp2,
    const float* __restrict__ pp1_b, const float* __restrict__ pp2_b,
    float* __restrict__ dout)
{
  __shared__ __align__(16) _Float16 X3[64][264];  // relu(total), [n][s]
  __shared__ __align__(16) _Float16 X4[64][264];  // relu(c1),    [n][s]

  const int b    = blockIdx.y;
  const int toff = blockIdx.x * 64;
  const int tid  = threadIdx.x;
  const int lane = tid & 31;
  const int wv   = tid >> 5;
  const float* tb = total + (size_t)b * 256 * kOutW;

  // Stage relu(total) as f16
  {
    const int n  = tid & 63;
    const int sq = tid >> 6;
    const int tg = toff + n;
    for (int j = 0; j < 64; ++j) {
      int s = sq * 64 + j;
      float v = 0.f;
      if (tg < kOutW) v = tb[(size_t)s * kOutW + tg];
      X3[n][s] = (_Float16)(v > 0.f ? v : 0.f);
    }
  }
  __syncthreads();

  // GEMM1: pp1 (256x256) x X3, + bias, relu -> X4
  for (int im = 0; im < 2; ++im) {
    const int mt = wv + im * 8;
    v8f acc[4] = {vzero8(), vzero8(), vzero8(), vzero8()};
    const _Float16* wrow = wp1 + (size_t)(mt * 16) * 256;
    for (int ks = 0; ks < 256; ks += 32) {
      v16h a = ld_frag(wrow + ks, 256, lane);
#pragma unroll
      for (int nt = 0; nt < 4; ++nt) {
        v16h bf = ld_frag(&X3[nt * 16][0] + ks, 264, lane);
        acc[nt] = wmma16(a, bf, acc[nt]);
      }
    }
    const int mbase = mt * 16 + 8 * (lane >> 4);
    const int ncol  = lane & 15;
#pragma unroll
    for (int v = 0; v < 8; ++v) {
      const float bias = pp1_b[mbase + v];
#pragma unroll
      for (int nt = 0; nt < 4; ++nt) {
        float c = acc[nt][v] + bias;
        X4[nt * 16 + ncol][mbase + v] = (_Float16)(c > 0.f ? c : 0.f);
      }
    }
  }
  __syncthreads();

  // GEMM2: pp2 (256x256) x X4, + bias -> d_out
  for (int im = 0; im < 2; ++im) {
    const int mt = wv + im * 8;
    v8f acc[4] = {vzero8(), vzero8(), vzero8(), vzero8()};
    const _Float16* wrow = wp2 + (size_t)(mt * 16) * 256;
    for (int ks = 0; ks < 256; ks += 32) {
      v16h a = ld_frag(wrow + ks, 256, lane);
#pragma unroll
      for (int nt = 0; nt < 4; ++nt) {
        v16h bf = ld_frag(&X4[nt * 16][0] + ks, 264, lane);
        acc[nt] = wmma16(a, bf, acc[nt]);
      }
    }
    const int mbase = mt * 16 + 8 * (lane >> 4);
    const int ncol  = lane & 15;
#pragma unroll
    for (int v = 0; v < 8; ++v) {
      const int q = mbase + v;
      const float bias = pp2_b[q];
#pragma unroll
      for (int nt = 0; nt < 4; ++nt) {
        int n = toff + nt * 16 + ncol;
        if (n < kOutW)
          dout[((size_t)b * 256 + q) * kOutW + n] = acc[nt][v] + bias;
      }
    }
  }
}

// ---------------------------------------------------------------------------
extern "C" void kernel_launch(void* const* d_in, const int* in_sizes, int n_in,
                              void* d_out, int out_size, void* d_ws, size_t ws_size,
                              hipStream_t stream)
{
  (void)in_sizes; (void)n_in; (void)out_size; (void)ws_size;

  const int*   x        = (const int*)  d_in[0];
  const float* causal_w = (const float*)d_in[1];
  const float* causal_b = (const float*)d_in[2];
  const float* filt_w   = (const float*)d_in[3];
  const float* filt_b   = (const float*)d_in[4];
  const float* gate_w   = (const float*)d_in[5];
  const float* gate_b   = (const float*)d_in[6];
  const float* dense_w  = (const float*)d_in[7];
  const float* dense_b  = (const float*)d_in[8];
  const float* skip_w   = (const float*)d_in[9];
  const float* skip_b   = (const float*)d_in[10];
  const float* pp1_w    = (const float*)d_in[11];
  const float* pp1_b    = (const float*)d_in[12];
  const float* pp2_w    = (const float*)d_in[13];
  const float* pp2_b    = (const float*)d_in[14];

  char* ws = (char*)d_ws;
  float*    h     = (float*)    (ws);
  _Float16* out16 = (_Float16*) (ws + H_BYTES);
  float*    total = (float*)    (ws + H_BYTES + O16_BYTES);
  _Float16* wfg   = (_Float16*) (ws + H_BYTES + O16_BYTES + TOT_BYTES);
  _Float16* w2    = wfg + WFG_HALF;
  _Float16* wp1   = w2  + W2_HALF;
  _Float16* wp2   = wp1 + WP_HALF;

  // skip accumulator must start at zero every call
  hipMemsetAsync(total, 0, TOT_BYTES, stream);

  // weight conversion (tiny, once per call to stay capture-deterministic)
  {
    const size_t ntot = WFG_HALF + W2_HALF + 2 * WP_HALF;
    k_cvt_weights<<<dim3((unsigned)((ntot + 255) / 256)), 256, 0, stream>>>(
        filt_w, gate_w, dense_w, skip_w, pp1_w, pp2_w, wfg, w2, wp1, wp2);
  }

  // causal conv (gather)
  k_causal<<<dim3((kT1 + 255) / 256, 64, kB), 256, 0, stream>>>(
      x, causal_w, causal_b, h);

  static const int dil[kL] = {1,2,4,8,16,32,64,128,256,512,
                              1,2,4,8,16,32,64,128,256,512};
  int off = 0;
  for (int i = 0; i < kL; ++i) {
    const int d        = dil[i];
    const int off_next = off + d;
    const int len      = kT1 - off_next;      // layer output length
    const int tail     = 2046 - off_next;     // skip starts at n == tail
    dim3 grid((len + 63) / 64, kB);
    k_gated<<<grid, 256, 0, stream>>>(h, wfg, filt_b, gate_b, out16,
                                      i, d, off, len);
    k_res_skip<<<grid, 256, 0, stream>>>(h, out16, w2, dense_b, skip_b, total,
                                         i, off_next, len, tail);
    off = off_next;
  }

  k_post<<<dim3((kOutW + 63) / 64, kB), 256, 0, stream>>>(
      total, wp1, wp2, pp1_b, pp2_b, (float*)d_out);
}
